// SpectraSentimentTokenModel_41077067219023
// MI455X (gfx1250) — matrix operand order repaired and checked by reading, not compiled, not run on hardware
//
#include <hip/hip_runtime.h>
#include <hip/hip_bf16.h>

// ---------------------------------------------------------------------------
// CDNA5 (gfx1250) SpectraSentimentTokenModel forward.
// GEMM1 on v_wmma_f32_16x16x32_bf16 (wave32), A via ds_load_b128 from
// double-buffered LDS, B via global_load_b128 from a pre-transposed bf16 W1.
// ---------------------------------------------------------------------------

typedef __attribute__((ext_vector_type(16))) __bf16 v16bf;
typedef __attribute__((ext_vector_type(8)))  __bf16 v8bf;
typedef __attribute__((ext_vector_type(8)))  float  v8f;

#define BB     32
#define TT     512
#define HID    768
#define MLPH   128
#define MROWS  (BB * TT)      // 16384
#define KSTEP  32
#define KITERS (HID / KSTEP)  // 24
#define MTILE  64             // rows of h per workgroup
#define XPAD   40             // row stride (80 B, multiple of 16 B)

__device__ __forceinline__ void store_bf16x4(__bf16* dst, float4 f) {
    union { unsigned long long u; __bf16 h[4]; } p;
    p.h[0] = (__bf16)f.x; p.h[1] = (__bf16)f.y;
    p.h[2] = (__bf16)f.z; p.h[3] = (__bf16)f.w;
    *reinterpret_cast<unsigned long long*>(dst) = p.u;   // ds_store_b64
}

__device__ __forceinline__ v16bf cat8(v8bf lo, v8bf hi) {
    return __builtin_shufflevector(lo, hi, 0,1,2,3,4,5,6,7,8,9,10,11,12,13,14,15);
}

// ---------------------------------------------------------------------------
// Kernel 0: W1T[n][k] = bf16(W1[k][n])   (one block per n, trivial cost)
// ---------------------------------------------------------------------------
__global__ __launch_bounds__(256)
void spectra_w1t_kernel(const float* __restrict__ W1, __bf16* __restrict__ W1T) {
    const int n = blockIdx.x;                       // 0..127
    for (int k = threadIdx.x; k < HID; k += 256)
        W1T[(size_t)n * HID + k] = (__bf16)W1[(size_t)k * MLPH + n];
}

// ---------------------------------------------------------------------------
// Kernel 1: h = relu(X @ W1 + b1);  logits = relu(h @ W2 + b2)   (fused)
//   256 blocks x 256 threads (8 waves); wave w owns N-tile [16w,16w+16).
// ---------------------------------------------------------------------------
__global__ __launch_bounds__(256)
void spectra_gemm_logits_kernel(const float* __restrict__ X,     // [16384,768]
                                const __bf16* __restrict__ W1T,  // [128,768] bf16
                                const float* __restrict__ b1,    // [128]
                                const float* __restrict__ W2,    // [128]
                                const float* __restrict__ b2,    // [1]
                                float* __restrict__ logits)      // [16384]
{
    __shared__ __align__(16) __bf16 Xs[2][MTILE][XPAD];  // double-buffered A tile
    __shared__ float Hs[MTILE][MLPH];                    // relu(h) tile

    const int tid  = threadIdx.x;
    const int lane = tid & 31;
    const int wave = tid >> 5;            // 0..7 -> N-tile
    const int half = lane >> 4;           // 0/1
    const int lrow = lane & 15;           // M (A) / N (B,C,D) within 16x16 tile
    const int m0   = blockIdx.x * MTILE;

    v8f acc[4];
#pragma unroll
    for (int mt = 0; mt < 4; ++mt)
#pragma unroll
        for (int r = 0; r < 8; ++r) acc[mt][r] = 0.0f;

    // per-wave base of this wave's 16 rows of W1T (n = wave*16 + lrow)
    const __bf16* wrow = W1T + (size_t)(wave * 16 + lrow) * HID + half * 16;

    int buf = 0;
    for (int kk = 0; kk < KITERS; ++kk) {
        const int K0 = kk * KSTEP;

        // ---- stage X tile 64x32 f32 -> bf16 (global_load_b128 + ds_store_b64)
#pragma unroll
        for (int e = tid; e < MTILE * (KSTEP / 4); e += 256) {   // 512 float4s
            const int row = e >> 3, q = e & 7;
            const float4* xg =
                (const float4*)(X + (size_t)(m0 + row) * HID + K0);
            store_bf16x4(&Xs[buf][row][q * 4], xg[q]);
        }
        if (kk + 1 < KITERS && tid < MTILE)
            __builtin_prefetch(&X[(size_t)(m0 + tid) * HID + K0 + KSTEP], 0, 0);
        __syncthreads();   // single barrier per iter (double buffering)

        // ---- B fragment: 16 contiguous bf16 from W1T (2x global_load_b128)
        //      layout: k = half*16 + j
        const v8bf blo = *(const v8bf*)(wrow + K0);
        const v8bf bhi = *(const v8bf*)(wrow + K0 + 8);
        const v16bf bfrag = cat8(blo, bhi);

        // ---- A fragments (layout k = half*16 + j ->
        //      one contiguous 16-element run -> 2x ds_load_b128) + WMMA
#pragma unroll
        for (int mt = 0; mt < 4; ++mt) {
            const __bf16* xrow = &Xs[buf][mt * 16 + lrow][half * 16];
            const v8bf alo = *(const v8bf*)(xrow);        // k = half*16 + 0..7
            const v8bf ahi = *(const v8bf*)(xrow + 8);    // k = half*16 + 8..15
            acc[mt] = __builtin_amdgcn_wmma_f32_16x16x32_bf16(
                false, cat8(alo, ahi), false, bfrag, (short)0, acc[mt],
                false, false);
        }
        buf ^= 1;
    }

    // bias + relu -> Hs  (C/D layout: M = r + 8*half, N = lane&15)
    {
        const int nn = wave * 16 + lrow;
        const float bias = b1[nn];
#pragma unroll
        for (int mt = 0; mt < 4; ++mt)
#pragma unroll
            for (int r = 0; r < 8; ++r)
                Hs[mt * 16 + r + 8 * half][nn] = fmaxf(acc[mt][r] + bias, 0.0f);
    }
    __syncthreads();

    // second GEMM (N=1): per-row dot with W2, then relu
    if (tid < MTILE) {
        float s = 0.0f;
#pragma unroll 8
        for (int c = 0; c < MLPH; ++c) s += Hs[tid][c] * W2[c];
        s += b2[0];
        logits[m0 + tid] = fmaxf(s, 0.0f);
    }
}

// ---------------------------------------------------------------------------
// Wave32 + cross-wave reductions; every thread returns the identical value.
// ---------------------------------------------------------------------------
__device__ __forceinline__ float wave_sum32(float v) {
#pragma unroll
    for (int off = 16; off > 0; off >>= 1) v += __shfl_xor(v, off, 32);
    return v;
}
__device__ __forceinline__ float wave_max32(float v) {
#pragma unroll
    for (int off = 16; off > 0; off >>= 1) v = fmaxf(v, __shfl_xor(v, off, 32));
    return v;
}
__device__ __forceinline__ float block_sum256(float v, float* red8) {
    v = wave_sum32(v);
    if ((threadIdx.x & 31) == 0) red8[threadIdx.x >> 5] = v;
    __syncthreads();
    const float s = ((red8[0] + red8[1]) + (red8[2] + red8[3])) +
                    ((red8[4] + red8[5]) + (red8[6] + red8[7]));
    __syncthreads();
    return s;
}
__device__ __forceinline__ float block_max256(float v, float* red8) {
    v = wave_max32(v);
    if ((threadIdx.x & 31) == 0) red8[threadIdx.x >> 5] = v;
    __syncthreads();
    const float s = fmaxf(fmaxf(fmaxf(red8[0], red8[1]), fmaxf(red8[2], red8[3])),
                          fmaxf(fmaxf(red8[4], red8[5]), fmaxf(red8[6], red8[7])));
    __syncthreads();
    return s;
}

// ---------------------------------------------------------------------------
// Kernel 2: budget projection (exact bisection, 100 iters), one block per b.
// ---------------------------------------------------------------------------
__global__ __launch_bounds__(256)
void spectra_solver_kernel(const float* __restrict__ logits,  // [B,T] relu'd
                           const int* __restrict__ mask,      // [B,T]
                           float* __restrict__ token_z)       // [B,T]
{
    __shared__ float red8[8];
    const int b = blockIdx.x, tid = threadIdx.x;
    const int t0 = tid, t1 = tid + 256;

    const int   mk0 = mask[b * TT + t0],  mk1 = mask[b * TT + t1];
    const float s0  = mk0 ? logits[b * TT + t0] : -1000.0f;
    const float s1  = mk1 ? logits[b * TT + t1] : -1000.0f;
    const float z0a = s0 * 10.0f;   // / TEMPERATURE (0.1)
    const float z0b = s1 * 10.0f;
    const float zca = fminf(fmaxf(z0a, 0.0f), 1.0f);
    const float zcb = fminf(fmaxf(z0b, 0.0f), 1.0f);

    const float f0     = block_sum256(zca + zcb, red8);
    const float length = block_sum256((float)mk0 + (float)mk1, red8);
    const float budget = rintf(0.3f * length);
    const float hi0    = fmaxf(block_max256(fmaxf(z0a, z0b), red8), 0.0f);

    float lo = 0.0f, hi = hi0;
    for (int it = 0; it < 100; ++it) {
        const float mid = 0.5f * (lo + hi);
        const float pa = fminf(fmaxf(z0a - mid, 0.0f), 1.0f);
        const float pb = fminf(fmaxf(z0b - mid, 0.0f), 1.0f);
        const float fm = block_sum256(pa + pb, red8);
        if (fm > budget) lo = mid; else hi = mid;
    }
    const float tau = 0.5f * (lo + hi);
    const bool keep = (f0 <= budget);
    token_z[b * TT + t0] = keep ? zca : fminf(fmaxf(z0a - tau, 0.0f), 1.0f);
    token_z[b * TT + t1] = keep ? zcb : fminf(fmaxf(z0b - tau, 0.0f), 1.0f);
}

// ---------------------------------------------------------------------------
// Kernel 3: z-weighted embedding pooling + linear classifier. One block per b.
// ---------------------------------------------------------------------------
__global__ __launch_bounds__(256)
void spectra_pool_cls_kernel(const float* __restrict__ token_z, // [B,T]
                             const int* __restrict__ mask,      // [B,T]
                             const int* __restrict__ ids,       // [B,T]
                             const float* __restrict__ embed,   // [V,768]
                             const float* __restrict__ Wc,      // [768,2]
                             const float* __restrict__ bc,      // [2]
                             float* __restrict__ cls)           // [B,2]
{
    __shared__ float wl[TT];
    __shared__ float pooled[HID];
    __shared__ float red8[8];
    const int b = blockIdx.x, tid = threadIdx.x;

    for (int t = tid; t < TT; t += 256)
        wl[t] = token_z[b * TT + t] * (float)mask[b * TT + t];
    __syncthreads();

    const float wsum = block_sum256(wl[tid] + wl[tid + 256], red8);

    float a0 = 0.0f, a1 = 0.0f, a2 = 0.0f;   // dims tid, tid+256, tid+512
    for (int t = 0; t < TT; ++t) {
        const float w = wl[t];
        const float* e = embed + (size_t)ids[b * TT + t] * HID;
        a0 += w * e[tid];
        a1 += w * e[tid + 256];
        a2 += w * e[tid + 512];
    }
    const float inv = 1.0f / (wsum + 1e-6f);
    pooled[tid]       = a0 * inv;
    pooled[tid + 256] = a1 * inv;
    pooled[tid + 512] = a2 * inv;
    __syncthreads();

    if (tid < 2) {
        float s = bc[tid];
        for (int d = 0; d < HID; ++d) s += pooled[d] * Wc[d * 2 + tid];
        cls[b * 2 + tid] = s;
    }
}

// ---------------------------------------------------------------------------
// Kernel 4: cross-entropy loss, one wave32.
// ---------------------------------------------------------------------------
__global__ __launch_bounds__(32)
void spectra_loss_kernel(const float* __restrict__ cls,     // [B,2]
                         const int* __restrict__ labels,    // [B]
                         float* __restrict__ loss)          // [1]
{
    const int b = threadIdx.x;
    const float x0 = cls[b * 2 + 0], x1 = cls[b * 2 + 1];
    const float mx = fmaxf(x0, x1);
    const float lse = mx + logf(expf(x0 - mx) + expf(x1 - mx));
    float li = lse - ((labels[b] == 0) ? x0 : x1);
#pragma unroll
    for (int off = 16; off > 0; off >>= 1)
        li += __shfl_down(li, off, 32);
    if (b == 0) loss[0] = li * (1.0f / (float)BB);
}

// ---------------------------------------------------------------------------
extern "C" void kernel_launch(void* const* d_in, const int* in_sizes, int n_in,
                              void* d_out, int out_size, void* d_ws, size_t ws_size,
                              hipStream_t stream) {
    const float* token_reps = (const float*)d_in[0];   // [32,512,768]
    const int*   attn_mask  = (const int*)  d_in[1];   // [32,512]
    const int*   input_ids  = (const int*)  d_in[2];   // [32,512]
    const int*   labels     = (const int*)  d_in[3];   // [32]
    const float* W1         = (const float*)d_in[4];   // [768,128]
    const float* b1         = (const float*)d_in[5];   // [128]
    const float* W2         = (const float*)d_in[6];   // [128,1]
    const float* b2         = (const float*)d_in[7];   // [1]
    const float* embed      = (const float*)d_in[8];   // [30522,768]
    const float* Wc         = (const float*)d_in[9];   // [768,2]
    const float* bc         = (const float*)d_in[10];  // [2]

    float* out      = (float*)d_out;
    float* cls_out  = out;            // [0,64)
    float* loss_out = out + 64;       // [64]
    float* token_z  = out + 65;       // [65, 65+16384)

    // workspace: logits [0, 64KB), W1T bf16 at byte 65536 (192 KB)
    float*  logits_ws = (float*)d_ws;
    __bf16* W1T       = (__bf16*)((char*)d_ws + (size_t)MROWS * sizeof(float));

    spectra_w1t_kernel<<<MLPH, 256, 0, stream>>>(W1, W1T);

    spectra_gemm_logits_kernel<<<MROWS / MTILE, 256, 0, stream>>>(
        token_reps, W1T, b1, W2, b2, logits_ws);

    spectra_solver_kernel<<<BB, 256, 0, stream>>>(logits_ws, attn_mask, token_z);

    spectra_pool_cls_kernel<<<BB, 256, 0, stream>>>(
        token_z, attn_mask, input_ids, embed, Wc, bc, cls_out);

    spectra_loss_kernel<<<1, 32, 0, stream>>>(cls_out, labels, loss_out);
}
